// TDTreeLSTM_27900107555184
// MI455X (gfx1250) — compile-verified, matrix operand order checked by reading, not run on hardware
//
#include <hip/hip_runtime.h>

// Tree-LSTM for MI455X (gfx1250): bf16 WMMA, wave32, async global->LDS gather.
// Phase 0: pack Wx/Wh fp32 -> bf16 WMMA B-fragment layout (one-shot).
// Phase 1: Xg2[t*B+b][j][gate] = x @ Wx^T + bias, bf16, gate-interleaved (b64 reads).
// Phase 2: persistent recurrence, 64 WGs = 8 batch-groups x 8 column-groups.
//          Recurrent weights REGISTER-resident (16 v16bf = 128 VGPRs per wave).
//          Parent-h gather: bf16 exchange buffer copied via
//          global_load_async_to_lds_b128 (ASYNCcnt) straight into LDS.
//          Column-groups sync per step via an L2 atomic counter barrier.

#define LSEQ 256
#define BATCH 128
#define DDIM 512
#define HDIM 512

typedef __attribute__((ext_vector_type(16))) __bf16 v16bf;
typedef __attribute__((ext_vector_type(8)))  __bf16 v8bf;
typedef __attribute__((ext_vector_type(8)))  float  v8f;

__device__ __forceinline__ unsigned short f2bf(float f) {
  union { float f; unsigned u; } v; v.f = f;
  unsigned u = v.u;
  u += 0x7FFFu + ((u >> 16) & 1u);   // round-to-nearest-even
  return (unsigned short)(u >> 16);
}
__device__ __forceinline__ float bf2f(unsigned short b) {
  union { unsigned u; float f; } v; v.u = ((unsigned)b) << 16;
  return v.f;
}
__device__ __forceinline__ float sigmoidf_(float x) { return 1.0f / (1.0f + __expf(-x)); }

// Per-lane K mapping (ISA 7.12.2, 16-bit operands): slots 0..7 -> K = half*8 + s,
// slots 8..15 -> K = 16 + half*8 + (s-8).
__device__ __forceinline__ v16bf load_frag_lds(const unsigned short* rowPtr, int kbase, int half) {
  union { v16bf v; v8bf h[2]; } u;
  u.h[0] = *(const v8bf*)(rowPtr + kbase + half * 8);
  u.h[1] = *(const v8bf*)(rowPtr + kbase + 16 + half * 8);
  return u.v;
}
__device__ __forceinline__ v16bf load_frag_g(const unsigned short* p) {
  union { v16bf v; v8bf h[2]; } u;
  u.h[0] = *(const v8bf*)(p);
  u.h[1] = *(const v8bf*)(p + 8);
  return u.v;
}

// ---------------- Phase 0: weight packing -------------------------------------
// frag = ((g*32 + ntile)*16 + ktile), 512 bf16 per frag (32 lanes x 16 slots).
__global__ __launch_bounds__(256)
void pack_weights(const float* __restrict__ W0, const float* __restrict__ W1,
                  const float* __restrict__ W2, const float* __restrict__ W3,
                  unsigned short* __restrict__ out) {
  int tid  = blockIdx.x * blockDim.x + threadIdx.x;  // 2^20 total
  int s    = tid & 15;
  int lane = (tid >> 4) & 31;
  int kt   = (tid >> 9) & 15;
  int nt   = (tid >> 13) & 31;
  int g    = tid >> 18;
  const float* W = (g == 0) ? W0 : (g == 1) ? W1 : (g == 2) ? W2 : W3;
  int n    = nt * 16 + (lane & 15);
  int half = lane >> 4;
  int k    = kt * 32 + ((s < 8) ? s : s + 8) + half * 8;
  out[tid] = f2bf(W[(size_t)n * 512 + k]);   // B[k][n] = W[n][k] (x @ W^T)
}

// ---------------- init: barrier counters + zero page --------------------------
__global__ __launch_bounds__(512)
void init_aux(unsigned* __restrict__ bar, unsigned short* __restrict__ zeroRow) {
  if (threadIdx.x < 8) bar[threadIdx.x] = 0u;
  zeroRow[threadIdx.x] = 0;   // 512 bf16 zeros
}

// ---------------- Phase 1: Xg2 = X @ Wx^T + bias (bf16, gate-interleaved) -----
__global__ __launch_bounds__(256)
void xgemm_kernel(const float* __restrict__ X, const unsigned short* __restrict__ Wp,
                  const float* __restrict__ bi_x, const float* __restrict__ bi_h,
                  const float* __restrict__ bf_x, const float* __restrict__ bf_h,
                  unsigned short* __restrict__ Xg2) {
  __shared__ unsigned short ldsA[128 * 40];   // 128 rows x 32 K, stride 40 (16B aligned)
  const int tid  = threadIdx.x;
  const int lane = tid & 31, wave = tid >> 5;
  const int half = lane >> 4, lcol = lane & 15;
  const int rowBase = blockIdx.x * 128;
  const int colBase = blockIdx.y * 64;        // 64 cols stay inside one gate

  v8f zero = {};
  v8f acc[4] = {zero, zero, zero, zero};
  const int gB   = colBase >> 9;
  const int ntg0 = (colBase & 511) >> 4;

  // Per-thread staging addressing: fixed column, rows stride by 8.
  const float* srcBase = X + (size_t)(rowBase + (tid >> 5)) * 512 + (tid & 31);
  unsigned short* ldsW = ldsA + (tid >> 5) * 40 + (tid & 31);

  for (int kt = 0; kt < 16; ++kt) {
    __syncthreads();
#pragma unroll
    for (int i = 0; i < 16; ++i)
      ldsW[i * 8 * 40] = f2bf(srcBase[kt * 32 + (size_t)i * 8 * 512]);
    __syncthreads();
    const unsigned short* rowPtr = ldsA + (wave * 16 + lcol) * 40;
    v16bf a = load_frag_lds(rowPtr, 0, half);
    v16bf bfr[4];
#pragma unroll
    for (int nt = 0; nt < 4; ++nt)
      bfr[nt] = load_frag_g(Wp + (((size_t)((gB * 32 + ntg0 + nt) * 16 + kt)) << 9) + lane * 16);
#pragma unroll
    for (int nt = 0; nt < 4; ++nt)
      acc[nt] = __builtin_amdgcn_wmma_f32_16x16x32_bf16(
          false, a, false, bfr[nt], (short)0, acc[nt], false, false);
  }
#pragma unroll
  for (int nt = 0; nt < 4; ++nt) {
    int n = colBase + nt * 16 + lcol;
    int g = n >> 9, j = n & 511;
    float bias = 0.0f;
    if (g == 0)      bias = bi_x[j] + bi_h[j];
    else if (g == 1) bias = bf_x[j] + bf_h[j];
#pragma unroll
    for (int e = 0; e < 8; ++e) {
      int row = rowBase + wave * 16 + e + half * 8;
      Xg2[((size_t)row * 512 + j) * 4 + g] = f2bf(acc[nt][e] + bias);
    }
  }
}

// ---------------- Phase 2: persistent recurrence ------------------------------
// 64 blocks x 512 thr (16 waves). Block = (batch-group bg, column-group cg):
// rows [bg*16, bg*16+16), hidden cols [cg*64, cg*64+64).
// Wave w owns N-tile (gate g = w>>2, hidden tile ht = w&3).
__global__ __launch_bounds__(512, 1)
void recur_kernel(const unsigned short* __restrict__ Xg2,
                  const unsigned short* __restrict__ Whp,
                  const int* __restrict__ parents,
                  float* __restrict__ Cbuf, float* __restrict__ Hbuf,
                  unsigned short* __restrict__ Hexch,
                  const unsigned short* __restrict__ zeroRow,
                  unsigned* __restrict__ bar) {
  __shared__ unsigned short ldsA[16 * 520];   // gathered parent h (bf16)
  __shared__ float ldsG[16 * 256];            // 16 N-tiles of gate values (16x16 f32)
  __shared__ int ldsP[16];
  const int tid  = threadIdx.x;
  const int lane = tid & 31, wave = tid >> 5;
  const int half = lane >> 4, lcol = lane & 15;
  const int bg = blockIdx.x >> 3, cg = blockIdx.x & 7;
  const int b0 = bg * 16, jc0 = cg * 64;
  const int g  = wave >> 2, ht = wave & 3;
  const int ntg = cg * 4 + ht;                // N-tile index within gate

  // One-time: weights for this wave's N-tile into registers (128 VGPRs).
  v16bf wreg[16];
#pragma unroll
  for (int kt = 0; kt < 16; ++kt)
    wreg[kt] = load_frag_g(Whp + (((size_t)((g * 32 + ntg) * 16 + kt)) << 9) + lane * 16);

  // Each thread owns one 32B gather chunk: row gr, cols [gc*16, gc*16+16).
  const int gr = tid >> 5, gc = tid & 31;
  const unsigned ldsDst = (unsigned)(size_t)(void*)(ldsA + gr * 520 + gc * 16);

  for (int t = 0; t < LSEQ; ++t) {
    if (tid < 16) ldsP[tid] = parents[t * BATCH + b0 + tid];
    __syncthreads();

    // Async gather of parent hidden state (bf16) straight into LDS.
    {
      int p = ldsP[gr];
      const unsigned short* src =
          (p >= 0) ? (Hexch + ((size_t)(b0 + gr) * LSEQ + p) * HDIM + gc * 16)
                   : (zeroRow + gc * 16);
      asm volatile(
          "global_load_async_to_lds_b128 %0, %1, off\n\t"
          "global_load_async_to_lds_b128 %0, %1, off offset:16"
          :: "v"(ldsDst), "v"(src) : "memory");
    }
    asm volatile("s_wait_asynccnt 0x0" ::: "memory");
    __syncthreads();

    // GEMM: two independent accumulator chains to break WMMA RAW serialization.
    v8f acc0 = {}, acc1 = {};
    const unsigned short* rowPtr = ldsA + lcol * 520;
#pragma unroll
    for (int kt = 0; kt < 16; kt += 2) {
      v16bf a0 = load_frag_lds(rowPtr, kt * 32, half);
      acc0 = __builtin_amdgcn_wmma_f32_16x16x32_bf16(
          false, a0, false, wreg[kt], (short)0, acc0, false, false);
      v16bf a1 = load_frag_lds(rowPtr, (kt + 1) * 32, half);
      acc1 = __builtin_amdgcn_wmma_f32_16x16x32_bf16(
          false, a1, false, wreg[kt + 1], (short)0, acc1, false, false);
    }
    v8f acc = acc0 + acc1;

    // Prefetch next step's Xg2 slice (address depends only on t -> full step of lead).
    if (t + 1 < LSEQ && tid < 128) {
      int r = tid >> 3, ch = tid & 7;
      const char* pf = (const char*)(Xg2 +
          ((size_t)((t + 1) * BATCH + b0 + r) * 512 + jc0) * 4) + ch * 64;
      __builtin_prefetch(pf, 0, 1);
    }

    // Exchange gate tiles through LDS (elementwise needs all 4 gates per col).
#pragma unroll
    for (int e = 0; e < 8; ++e)
      ldsG[wave * 256 + (e + 8 * half) * 16 + lcol] = acc[e];
    __syncthreads();

    // Elementwise LSTM update for this WG's 16 rows x 64 hidden cols.
    for (int idx = tid; idx < 16 * 64; idx += 512) {
      int m = idx >> 6, jj = idx & 63;
      int htl = jj >> 4, col = jj & 15;
      int mb16 = m * 16 + col;
      float gi = ldsG[(0 * 4 + htl) * 256 + mb16];
      float gf = ldsG[(1 * 4 + htl) * 256 + mb16];
      float go = ldsG[(2 * 4 + htl) * 256 + mb16];
      float gu = ldsG[(3 * 4 + htl) * 256 + mb16];
      int b = b0 + m;
      int j = jc0 + jj;
      const unsigned short* xp = Xg2 + ((size_t)(t * BATCH + b) * 512 + j) * 4;
      gi += bf2f(xp[0]);
      gf += bf2f(xp[1]);
      go += bf2f(xp[2]);
      gu += bf2f(xp[3]);
      int p = ldsP[m];
      float pc = (p >= 0) ? Cbuf[((size_t)b * LSEQ + p) * HDIM + j] : 0.0f;
      float c = sigmoidf_(gi) * tanhf(gu) + sigmoidf_(gf) * pc;
      float h = sigmoidf_(go) * tanhf(c);
      size_t o = ((size_t)b * LSEQ + t) * HDIM + j;
      Cbuf[o]  = c;
      Hbuf[o]  = h;          // fp32 output [B, L, H]
      Hexch[o] = f2bf(h);    // bf16 exchange copy for next-step gathers
    }

    // Inter-WG barrier across the 8 column-groups of this batch-group.
    __threadfence();             // release this WG's h/c to agent scope
    __syncthreads();             // all threads' fences done before signal
    if (t != LSEQ - 1) {
      if (tid == 0) {
        __hip_atomic_fetch_add(bar + bg, 1u, __ATOMIC_RELEASE, __HIP_MEMORY_SCOPE_AGENT);
        unsigned target = 8u * (unsigned)(t + 1);
        while (__hip_atomic_load(bar + bg, __ATOMIC_ACQUIRE, __HIP_MEMORY_SCOPE_AGENT) < target)
          __builtin_amdgcn_s_sleep(2);
      }
      __syncthreads();           // release spinners' result to whole WG
      __threadfence();           // acquire: invalidate WGP-level caches before gather
    }
  }
}

// ---------------- Launch -------------------------------------------------------
extern "C" void kernel_launch(void* const* d_in, const int* in_sizes, int n_in,
                              void* d_out, int out_size, void* d_ws, size_t ws_size,
                              hipStream_t stream) {
  const float* X    = (const float*)d_in[0];
  const float* Wi_x = (const float*)d_in[1];
  const float* bi_x = (const float*)d_in[2];
  const float* Wi_h = (const float*)d_in[3];
  const float* bi_h = (const float*)d_in[4];
  const float* Wf_x = (const float*)d_in[5];
  const float* bf_x = (const float*)d_in[6];
  const float* Wf_h = (const float*)d_in[7];
  const float* bf_h = (const float*)d_in[8];
  const float* Wo_x = (const float*)d_in[9];
  const float* Wo_h = (const float*)d_in[10];
  const float* Wu_x = (const float*)d_in[11];
  const float* Wu_h = (const float*)d_in[12];
  const int* parents = (const int*)d_in[13];

  char* ws = (char*)d_ws;
  unsigned short* WxP = (unsigned short*)(ws);                         // 2 MB
  unsigned short* WhP = (unsigned short*)(ws + ((size_t)2 << 20));     // 2 MB
  unsigned short* Xg2 = (unsigned short*)(ws + ((size_t)4 << 20));     // 128 MB bf16
  size_t xgBytes = (size_t)LSEQ * BATCH * 2048 * sizeof(unsigned short);
  char* p = ws + ((size_t)4 << 20) + xgBytes;
  float* Cbuf = (float*)p;                                             // 64 MB
  p += (size_t)BATCH * LSEQ * HDIM * sizeof(float);
  unsigned short* Hexch = (unsigned short*)p;                          // 32 MB
  p += (size_t)BATCH * LSEQ * HDIM * sizeof(unsigned short);
  unsigned* bar = (unsigned*)p;                                        // 32 B
  unsigned short* zeroRow = (unsigned short*)(p + 1024);               // 1 KB
  float* Hbuf = (float*)d_out;

  pack_weights<<<4096, 256, 0, stream>>>(Wi_x, Wf_x, Wo_x, Wu_x, WxP);
  pack_weights<<<4096, 256, 0, stream>>>(Wi_h, Wf_h, Wo_h, Wu_h, WhP);
  init_aux<<<1, 512, 0, stream>>>(bar, zeroRow);
  xgemm_kernel<<<dim3(256, 32), 256, 0, stream>>>(X, WxP, bi_x, bi_h, bf_x, bf_h, Xg2);
  recur_kernel<<<64, 512, 0, stream>>>(Xg2, WhP, parents, Cbuf, Hbuf, Hexch, zeroRow, bar);
}
